// SAGE_agg1_30081950941676
// MI455X (gfx1250) — matrix operand order repaired and checked by reading, not compiled, not run on hardware
//
#include <hip/hip_runtime.h>
#include <hip/hip_bf16.h>

// ---------------------------------------------------------------------------
// GraphSAGE 2-layer forward for MI455X (gfx1250, wave32).
//  - scatter-mean aggregation via global_atomic_add_f32 (L2-resident)
//  - activations pre-converted to bf16 once; fused dual-GEMM
//    (agg@Wl + x@Wr) is pure global_load_b128 + v_wmma_f32_16x16x32_bf16,
//    register-blocked over 4 column tiles per wave for 4x A-tile reuse
//  - in-place log-softmax with wave shuffles
// ---------------------------------------------------------------------------

#define N_NODES   50000
#define N_EDGES   600000
#define IN_CH     128
#define HID_CH    128
#define N_CLASSES 64
#define NT        4            // column tiles per wave (16*NT = 64 columns)

typedef __attribute__((ext_vector_type(16))) __bf16         v16bf;
typedef __attribute__((ext_vector_type(16))) unsigned short v16u;
typedef __attribute__((ext_vector_type(8)))  float          v8f;

__device__ __forceinline__ unsigned short f2bf(float f) {
    unsigned int u = __float_as_uint(f);
    unsigned int r = u + 0x7FFFu + ((u >> 16) & 1u);   // round-to-nearest-even
    return (unsigned short)(r >> 16);
}

// ---------------- utility kernels ------------------------------------------

__global__ void sage_zero_kernel(float* __restrict__ p, int n) {
    int i = blockIdx.x * blockDim.x + threadIdx.x;
    if (i < n) p[i] = 0.0f;
}

// elementwise f32 -> bf16
__global__ void sage_cvt_kernel(const float* __restrict__ in,
                                unsigned short* __restrict__ out, int n) {
    int i = blockIdx.x * blockDim.x + threadIdx.x;
    if (i < n) out[i] = f2bf(in[i]);
}

// W[K][N] (f32, row-major) -> WT[N][K] (bf16, row-major), K == 128
__global__ void sage_wt_kernel(const float* __restrict__ W,
                               unsigned short* __restrict__ WT, int N) {
    int idx = blockIdx.x * blockDim.x + threadIdx.x;
    if (idx >= 128 * N) return;
    int k = idx / N, n = idx % N;
    WT[(size_t)n * 128 + k] = f2bf(W[(size_t)k * N + n]);
}

__global__ void sage_deg_kernel(const long long* __restrict__ dst,
                                float* __restrict__ deg, int n_edges) {
    int e = blockIdx.x * blockDim.x + threadIdx.x;
    if (e < n_edges) atomicAdd(&deg[(int)dst[e]], 1.0f);
}

// One wave per edge: lane l handles channels l, l+32, l+64, l+96 (coalesced).
__global__ void sage_agg_kernel(const float* __restrict__ X,
                                const long long* __restrict__ src,
                                const long long* __restrict__ dst,
                                float* __restrict__ agg, int n_edges) {
    int wave = blockIdx.x * (blockDim.x >> 5) + (threadIdx.x >> 5);
    if (wave >= n_edges) return;
    int lane = threadIdx.x & 31;
    const float* xs = X   + (size_t)src[wave] * IN_CH;
    float*       ad = agg + (size_t)dst[wave] * IN_CH;
#pragma unroll
    for (int j = 0; j < 4; ++j) {
        int c = lane + 32 * j;
        atomicAdd(&ad[c], xs[c]);
    }
}

// aggB[row][c] = bf16(agg[row][c] / max(deg[row],1))  (block = row, 128 thr)
__global__ void sage_norm_kernel(const float* __restrict__ agg,
                                 const float* __restrict__ deg,
                                 unsigned short* __restrict__ aggB) {
    int row = blockIdx.x;
    float inv = 1.0f / fmaxf(deg[row], 1.0f);
    size_t i = (size_t)row * IN_CH + threadIdx.x;
    aggB[i] = f2bf(agg[i] * inv);
}

// ---------------- WMMA GEMM -------------------------------------------------
// A-tile (16x32 bf16) per ISA layout: lane = {half = lane>>4, row = lane&15};
// elems 0..7  = A[row][k0 + 8*half + i], elems 8..15 = A[row][k0+16+8*half+i].
// A stored row-major bf16 [M][128]  -> two 16B (b128) loads, 32B apart.
__device__ __forceinline__ v16bf load_a16(const unsigned short* __restrict__ A,
                                          int row, int half, int k0) {
    const unsigned short* p = A + (size_t)row * 128 + k0 + 8 * half;
    struct U { uint4 a, b; } t = { *(const uint4*)p, *(const uint4*)(p + 16) };
    return __builtin_bit_cast(v16bf, __builtin_bit_cast(v16u, t));
}

// B-tile (32x16 bf16): lane n = lane&15 (column); elems 0..7 = K k0+8h+i, elems 8..15 = K k0+16+8h+i.
// WT stored [N][128] bf16 -> two 16B (b128) loads, 32B apart.
__device__ __forceinline__ v16bf load_b16(const unsigned short* __restrict__ WT,
                                          int n, int half, int k0) {
    const unsigned short* p = WT + (size_t)n * 128 + k0 + 8 * half;
    struct U { uint4 a, b; } t = { *(const uint4*)p, *(const uint4*)(p + 16) };
    return __builtin_bit_cast(v16bf, __builtin_bit_cast(v16u, t));
}

// One wave computes a 16 x (16*NT) stripe of  out = Aagg@Wl + Aself@Wr + bias
// (K = 128).  A-tiles are loaded once per K-step and reused across NT column
// tiles (4x L2 traffic reduction).  Optionally emits a bf16 copy of the
// (post-ReLU) output for the next layer.
__global__ __launch_bounds__(32) void sage_gemm_kernel(
    const unsigned short* __restrict__ Aagg, const unsigned short* __restrict__ Aself,
    const unsigned short* __restrict__ WTl,  const unsigned short* __restrict__ WTr,
    const float* __restrict__ bias, float* __restrict__ out,
    unsigned short* __restrict__ outB, int N, int relu) {
    int tm  = blockIdx.x;
    int tn0 = blockIdx.y * NT;            // first column tile of this wave
    int lane = threadIdx.x;
    int r    = lane & 15;
    int half = lane >> 4;
    int row  = tm * 16 + r;               // A row for this lane
    v8f acc[NT] = {};
#pragma unroll
    for (int kt = 0; kt < 4; ++kt) {
        int k0 = kt * 32;
        v16bf a0 = load_a16(Aagg,  row, half, k0);
        v16bf a1 = load_a16(Aself, row, half, k0);
#pragma unroll
        for (int j = 0; j < NT; ++j) {
            int n = (tn0 + j) * 16 + r;   // B column for this lane
            v16bf b0 = load_b16(WTl, n, half, k0);
            acc[j] = __builtin_amdgcn_wmma_f32_16x16x32_bf16(
                false, a0, false, b0, (short)0, acc[j], false, false);
            v16bf b1 = load_b16(WTr, n, half, k0);
            acc[j] = __builtin_amdgcn_wmma_f32_16x16x32_bf16(
                false, a1, false, b1, (short)0, acc[j], false, false);
        }
    }
#pragma unroll
    for (int j = 0; j < NT; ++j) {
        int n = (tn0 + j) * 16 + r;
        float bn = bias[n];
#pragma unroll
        for (int v = 0; v < 8; ++v) {     // C/D layout: M = v + 8*half
            int m = tm * 16 + v + 8 * half;
            float val = acc[j][v] + bn;
            if (relu) val = fmaxf(val, 0.0f);
            size_t o = (size_t)m * N + n;
            out[o] = val;
            if (outB) outB[o] = f2bf(val);
        }
    }
}

// ---------------- log-softmax (in place, C = 64, one wave per row) ----------
__global__ __launch_bounds__(32) void sage_lsm_kernel(float* __restrict__ out) {
    int lane = threadIdx.x;
    float* p = out + (size_t)blockIdx.x * N_CLASSES;
    float a = p[lane], b = p[lane + 32];
    float m = fmaxf(a, b);
#pragma unroll
    for (int off = 16; off > 0; off >>= 1) m = fmaxf(m, __shfl_xor(m, off, 32));
    float s = expf(a - m) + expf(b - m);
#pragma unroll
    for (int off = 16; off > 0; off >>= 1) s += __shfl_xor(s, off, 32);
    float lse = m + logf(s);
    p[lane]      = a - lse;
    p[lane + 32] = b - lse;
}

// ---------------------------------------------------------------------------

extern "C" void kernel_launch(void* const* d_in, const int* in_sizes, int n_in,
                              void* d_out, int out_size, void* d_ws, size_t ws_size,
                              hipStream_t stream) {
    const float*     x    = (const float*)d_in[0];
    const long long* ei   = (const long long*)d_in[1];   // int64 (2, E)
    const float*     Wl1  = (const float*)d_in[2];
    const float*     bl1  = (const float*)d_in[3];
    const float*     Wr1  = (const float*)d_in[4];
    const float*     Wl2  = (const float*)d_in[5];
    const float*     bl2  = (const float*)d_in[6];
    const float*     Wr2  = (const float*)d_in[7];
    const long long* src  = ei;
    const long long* dst  = ei + N_EDGES;
    float* out = (float*)d_out;

    // scratch carve-out (256B aligned)
    char*  ws  = (char*)d_ws;
    size_t off = 0;
    auto carve = [&](size_t bytes) {
        size_t cur = off;
        off = (off + bytes + 255) & ~(size_t)255;
        return cur;
    };
    const size_t FEAT = (size_t)N_NODES * IN_CH;
    float*          deg  = (float*)(ws + carve((size_t)N_NODES * 4));
    float*          agg  = (float*)(ws + carve(FEAT * 4));   // f32 atomic accum
    float*          h    = (float*)(ws + carve(FEAT * 4));   // f32 hidden (for agg2)
    unsigned short* xB   = (unsigned short*)(ws + carve(FEAT * 2));
    unsigned short* aggB = (unsigned short*)(ws + carve(FEAT * 2));
    unsigned short* hB   = (unsigned short*)(ws + carve(FEAT * 2));
    unsigned short* WTl1 = (unsigned short*)(ws + carve((size_t)HID_CH * 128 * 2));
    unsigned short* WTr1 = (unsigned short*)(ws + carve((size_t)HID_CH * 128 * 2));
    unsigned short* WTl2 = (unsigned short*)(ws + carve((size_t)N_CLASSES * 128 * 2));
    unsigned short* WTr2 = (unsigned short*)(ws + carve((size_t)N_CLASSES * 128 * 2));

    const int AGG_ELEMS = (int)FEAT;

    // weight transpose + bf16 convert (tiny); x -> bf16 once
    sage_wt_kernel<<<(128 * HID_CH + 255) / 256, 256, 0, stream>>>(Wl1, WTl1, HID_CH);
    sage_wt_kernel<<<(128 * HID_CH + 255) / 256, 256, 0, stream>>>(Wr1, WTr1, HID_CH);
    sage_wt_kernel<<<(128 * N_CLASSES + 255) / 256, 256, 0, stream>>>(Wl2, WTl2, N_CLASSES);
    sage_wt_kernel<<<(128 * N_CLASSES + 255) / 256, 256, 0, stream>>>(Wr2, WTr2, N_CLASSES);
    sage_cvt_kernel<<<(AGG_ELEMS + 255) / 256, 256, 0, stream>>>(x, xB, AGG_ELEMS);

    // degree (shared by both layers)
    sage_zero_kernel<<<(N_NODES + 255) / 256, 256, 0, stream>>>(deg, N_NODES);
    sage_deg_kernel<<<(N_EDGES + 255) / 256, 256, 0, stream>>>(dst, deg, N_EDGES);

    // ---- layer 1 ----
    sage_zero_kernel<<<(AGG_ELEMS + 255) / 256, 256, 0, stream>>>(agg, AGG_ELEMS);
    sage_agg_kernel<<<(N_EDGES + 7) / 8, 256, 0, stream>>>(x, src, dst, agg, N_EDGES);
    sage_norm_kernel<<<N_NODES, 128, 0, stream>>>(agg, deg, aggB);
    sage_gemm_kernel<<<dim3(N_NODES / 16, HID_CH / (16 * NT)), 32, 0, stream>>>(
        aggB, xB, WTl1, WTr1, bl1, h, hB, HID_CH, /*relu=*/1);

    // ---- layer 2 ----
    sage_zero_kernel<<<(AGG_ELEMS + 255) / 256, 256, 0, stream>>>(agg, AGG_ELEMS);
    sage_agg_kernel<<<(N_EDGES + 7) / 8, 256, 0, stream>>>(h, src, dst, agg, N_EDGES);
    sage_norm_kernel<<<N_NODES, 128, 0, stream>>>(agg, deg, aggB);
    sage_gemm_kernel<<<dim3(N_NODES / 16, N_CLASSES / (16 * NT)), 32, 0, stream>>>(
        aggB, hB, WTl2, WTr2, bl2, out, (unsigned short*)nullptr, N_CLASSES, /*relu=*/0);

    // log-softmax in place on d_out
    sage_lsm_kernel<<<N_NODES, 32, 0, stream>>>(out);
}
